// VONet_68985764708951
// MI455X (gfx1250) — matrix operand order repaired and checked by reading, not compiled, not run on hardware
//
#include <hip/hip_runtime.h>

// ---------------------------------------------------------------------------
// Shapes / constants
// ---------------------------------------------------------------------------
constexpr int Nn    = 8;
constexpr int Cc    = 384;
constexpr int Hh    = 96;
constexpr int Ww    = 128;
constexpr int HWc   = Hh * Ww;              // 12288
constexpr int HALFC = Cc / 2;               // 192
constexpr long Etot = (long)Nn * Cc * HWc;  // 37,748,736 elements per tensor
constexpr float EPSv = 1e-5f;

typedef __bf16 bhalf;
typedef bhalf v16bf __attribute__((ext_vector_type(16)));
typedef float v8f   __attribute__((ext_vector_type(8)));

__device__ __forceinline__ float bf2f(unsigned short u) {
    unsigned int x = ((unsigned int)u) << 16;
    float f; __builtin_memcpy(&f, &x, 4); return f;
}
__device__ __forceinline__ unsigned short f2bf(float f) {
    unsigned int u; __builtin_memcpy(&u, &f, 4);
    u += 0x7fffu + ((u >> 16) & 1u);        // round-to-nearest-even
    return (unsigned short)(u >> 16);
}

// ---------------------------------------------------------------------------
// Tensor Data Mover support (gfx1250). Guarded so compile can never be lost.
// ---------------------------------------------------------------------------
#if defined(__has_builtin)
#if __has_builtin(__builtin_amdgcn_tensor_load_to_lds) && __has_builtin(__builtin_amdgcn_s_wait_tensorcnt)
#define USE_TDM 1
#endif
#endif
#ifndef USE_TDM
#define USE_TDM 0
#endif

constexpr int BM  = 128;
constexpr int BNT = 128;
constexpr int BK  = 32;
constexpr int LDP = 40;   // padded LDS row stride (ushorts): 64B data + 16B pad

#if USE_TDM
typedef unsigned int v4u_t __attribute__((ext_vector_type(4)));
typedef int          v8i_t __attribute__((ext_vector_type(8)));
typedef int          v4i_t __attribute__((ext_vector_type(4)));

// Async-load a BM x BK bf16 weight tile (row pitch = rowStride elements) into
// LDS at lds_byte_addr, laid out with an 80-byte padded row stride:
//   D# pad: interval = 16 DWORDs (one 64B row), amount = 4 DWORDs (16B).
__device__ __forceinline__ void tdm_load_tile(unsigned lds_byte_addr,
                                              const unsigned short* gsrc,
                                              int rowStride) {
    unsigned long long ga = (unsigned long long)(uintptr_t)gsrc;
    v4u_t g0;
    g0.x = 1u;                                           // count=1 (valid, user)
    g0.y = lds_byte_addr;                                // lds_addr [63:32]
    g0.z = (unsigned)(ga & 0xffffffffu);                 // global_addr [95:64]
    g0.w = (unsigned)((ga >> 32) & 0x1ffffffu)           // global_addr [120:96]
         | (2u << 30);                                   // type=2 ("image")
    v8i_t g1;
    g1[0] = (1 << 16)                                    // data_size = 2 bytes
          | (1 << 20)                                    // pad_enable
          | (3 << 22)                                    // pad_interval: 16 DW
          | (3 << 25);                                   // pad_amount:   4 DW
    g1[1] = (BK & 0xffff) << 16;                         // tensor_dim0 = BK
    g1[2] = ((BM & 0xffff) << 16);                       // tensor_dim1 = BM
    g1[3] = (BK << 16);                                  // tile_dim0   = BK
    g1[4] = BM;                                          // tile_dim1   = BM
    g1[5] = rowStride;                                   // dim0_stride lo32
    g1[6] = 0;                                           // dim0_stride hi / dim1_stride lo
    g1[7] = 0;
    v4i_t z4 = {0, 0, 0, 0};
#if __clang_major__ >= 23
    v8i_t z8 = {0, 0, 0, 0, 0, 0, 0, 0};
    __builtin_amdgcn_tensor_load_to_lds(g0, g1, z4, z4, z8, 0);
#else
    __builtin_amdgcn_tensor_load_to_lds(g0, g1, z4, z4, 0);
#endif
}
#endif // USE_TDM

// ---------------------------------------------------------------------------
// f32 -> bf16 conversion (grid-stride)
// ---------------------------------------------------------------------------
__global__ void cvt_f32_bf16(const float* __restrict__ in,
                             unsigned short* __restrict__ out, long n) {
    long stride = (long)gridDim.x * blockDim.x;
    for (long i = (long)blockIdx.x * blockDim.x + threadIdx.x; i < n; i += stride)
        out[i] = f2bf(in[i]);
}

// ---------------------------------------------------------------------------
// Implicit-GEMM conv (NCHW, OIHW weights flattened to [co][Ktot] bf16).
//   out[co][p] = sum_k  W[co][k] * act(k, p)
// BN affine + channel-shuffle folded into the activation fetch:
//   ci < HALFC -> actLo/scLo/shLo,  else actHi/scHi/shHi.
// Tiles: BM=128 x BNT=128 x BK=32, 256 threads = 8 waves (2x4).
// Each wave: 4x2 tiles of v_wmma_f32_16x16x32_bf16, f32 accumulate.
// Weight tiles arrive via the Tensor Data Mover (wave 0 issues, TENSORcnt).
// ---------------------------------------------------------------------------
template<int KSZ, bool AFFINE>
__global__ __launch_bounds__(256, 1)
void gemm_conv(const unsigned short* __restrict__ actLo,
               const unsigned short* __restrict__ actHi,
               const float* __restrict__ scLo, const float* __restrict__ shLo,
               const float* __restrict__ scHi, const float* __restrict__ shHi,
               const unsigned short* __restrict__ wbf,
               unsigned short* __restrict__ outRaw)
{
    constexpr int Ktot = Cc * KSZ * KSZ;
    constexpr int KT   = Ktot / BK;

    __shared__ unsigned short As[2][BM * LDP];   // weights  [row][k]
    __shared__ unsigned short Bs[2][BNT * LDP];  // act      [pos][k]

    const int tid    = threadIdx.x;
    const int pBase  = blockIdx.x * BNT;
    const int coBase = blockIdx.y * BM;
    const int nimg   = pBase / HWc;          // uniform per block
    const int qBase  = pBase - nimg * HWc;

    // ---- B (activation) staging -----------------------------------------
    const int bPos = tid & 127;              // spatial within tile
    const int kSub = tid >> 7;               // 0/1: even/odd k
    const int q    = qBase + bPos;
    const int h0   = q >> 7;
    const int w0   = q & 127;
    float bReg[16];
    unsigned short bRegU[16];

    auto gloadB = [&](int kBase) {
        if constexpr (KSZ == 1 && !AFFINE) {
            // pure bf16 copy: no bounds / affine / conversion
            const unsigned short* src =
                actLo + (nimg * Cc + kBase) * HWc + qBase + bPos;
            #pragma unroll
            for (int i = 0; i < 16; ++i)
                bRegU[i] = src[((i << 1) + kSub) * HWc];
        } else {
            #pragma unroll
            for (int i = 0; i < 16; ++i) {
                int k  = (i << 1) + kSub;
                int kg = kBase + k;
                int ci, r, s;
                if constexpr (KSZ == 1) { ci = kg; r = 0; s = 0; }
                else { ci = kg / 9; int rs = kg - ci * 9; int rr = rs / 3;
                       r = rr - 1; s = rs - rr * 3 - 1; }
                int hh = h0 + r, ww2 = w0 + s;
                float v = 0.f;
                if ((unsigned)hh < (unsigned)Hh && (unsigned)ww2 < (unsigned)Ww) {
                    bool lo = ci < HALFC;
                    const unsigned short* src = lo ? actLo : actHi;
                    float x = bf2f(src[(nimg * Cc + ci) * HWc + (hh << 7) + ww2]);
                    if constexpr (AFFINE) {
                        const float* sc = lo ? scLo : scHi;
                        const float* sh = lo ? shLo : shHi;
                        x = x * sc[ci] + sh[ci];     // zero-pad stays zero
                    }
                    v = x;
                }
                bReg[i] = v;
            }
        }
    };

    auto lstoreB = [&](int buf) {
        if constexpr (KSZ == 1 && !AFFINE) {
            #pragma unroll
            for (int i = 0; i < 16; ++i)
                Bs[buf][bPos * LDP + (i << 1) + kSub] = bRegU[i];
        } else {
            #pragma unroll
            for (int i = 0; i < 16; ++i)
                Bs[buf][bPos * LDP + (i << 1) + kSub] = f2bf(bReg[i]);
        }
    };

#if !USE_TDM
    // ---- fallback A (weight) staging: 256 thr * 32B = 8KB tile ----------
    const int aRow = tid >> 1;
    const int aCol = (tid & 1) * 16;
    const unsigned short* aPtr = wbf + (coBase + aRow) * Ktot + aCol;
    uint4 aReg0, aReg1;
    auto gloadA = [&](int kBase) {
        aReg0 = *(const uint4*)(aPtr + kBase);
        aReg1 = *(const uint4*)(aPtr + kBase + 8);
        if (kBase + BK < Ktot) __builtin_prefetch(aPtr + kBase + BK, 0, 3);
    };
    auto lstoreA = [&](int buf) {
        *(uint4*)&As[buf][aRow * LDP + aCol]     = aReg0;
        *(uint4*)&As[buf][aRow * LDP + aCol + 8] = aReg1;
    };
#endif

    const int lane = tid & 31;
    const int wave = tid >> 5;
    const int wm = wave >> 2, wn = wave & 3;   // 2 x 4 wave grid
    const int lm = lane & 15, lh = lane >> 4;

    v8f acc[4][2];
    #pragma unroll
    for (int mt = 0; mt < 4; ++mt)
        #pragma unroll
        for (int nt = 0; nt < 2; ++nt) {
            v8f z = {0.f,0.f,0.f,0.f,0.f,0.f,0.f,0.f};
            acc[mt][nt] = z;
        }

    auto math = [&](int buf) {
        union FragU { uint4 u[2]; v16bf v; } af[4], bq[2];
        #pragma unroll
        for (int mt = 0; mt < 4; ++mt) {
            // A frag (16x32 bf16): lane lm row, K runs {8*lh..+8} and {+16}
            int row = wm * 64 + mt * 16 + lm;
            af[mt].u[0] = *(const uint4*)&As[buf][row * LDP + lh * 8];
            af[mt].u[1] = *(const uint4*)&As[buf][row * LDP + lh * 8 + 16];
        }
        #pragma unroll
        for (int nt = 0; nt < 2; ++nt) {
            // B frag (32x16 bf16): lane lm col, K runs {8*lh..+8} and {+16}
            int col = wn * 32 + nt * 16 + lm;
            bq[nt].u[0] = *(const uint4*)&Bs[buf][col * LDP + lh * 8];
            bq[nt].u[1] = *(const uint4*)&Bs[buf][col * LDP + lh * 8 + 16];
        }
        #pragma unroll
        for (int mt = 0; mt < 4; ++mt)
            #pragma unroll
            for (int nt = 0; nt < 2; ++nt)
                acc[mt][nt] = __builtin_amdgcn_wmma_f32_16x16x32_bf16(
                    false, af[mt].v, false, bq[nt].v,
                    (short)0, acc[mt][nt], false, false);
    };

    // ---- main loop: LDS double-buffered ---------------------------------
#if USE_TDM
    const unsigned short* aTileBase = wbf + coBase * Ktot;
    const unsigned ldsA0 = (unsigned)(uintptr_t)&As[0][0];
    const unsigned ldsA1 = (unsigned)(uintptr_t)&As[1][0];
    const bool isWave0 = (tid >> 5) == 0;

    if (isWave0) tdm_load_tile(ldsA0, aTileBase, Ktot);
    gloadB(0);
    lstoreB(0);
    if (isWave0) __builtin_amdgcn_s_wait_tensorcnt(0);
    __syncthreads();
    for (int kt = 0; kt < KT; ++kt) {
        if (kt + 1 < KT) {
            if (isWave0)
                tdm_load_tile(((kt + 1) & 1) ? ldsA1 : ldsA0,
                              aTileBase + (kt + 1) * BK, Ktot);
            gloadB((kt + 1) * BK);
        }
        math(kt & 1);
        if (kt + 1 < KT) {
            lstoreB((kt + 1) & 1);
            if (isWave0) __builtin_amdgcn_s_wait_tensorcnt(0);
            __syncthreads();
        }
    }
#else
    gloadA(0); gloadB(0);
    lstoreA(0); lstoreB(0);
    __syncthreads();
    for (int kt = 0; kt < KT; ++kt) {
        if (kt + 1 < KT) { gloadA((kt + 1) * BK); gloadB((kt + 1) * BK); }
        math(kt & 1);
        if (kt + 1 < KT) {
            lstoreA((kt + 1) & 1); lstoreB((kt + 1) & 1);
            __syncthreads();
        }
    }
#endif

    // ---- epilogue: f32 acc -> bf16 raw conv output ----------------------
    const int base = nimg * Cc * HWc;
    #pragma unroll
    for (int mt = 0; mt < 4; ++mt) {
        int co0 = coBase + wm * 64 + mt * 16 + lh * 8;
        #pragma unroll
        for (int nt = 0; nt < 2; ++nt) {
            int pc = qBase + wn * 32 + nt * 16 + lm;
            #pragma unroll
            for (int j = 0; j < 8; ++j)
                outRaw[base + (co0 + j) * HWc + pc] = f2bf(acc[mt][nt][j]);
        }
    }
}

// ---------------------------------------------------------------------------
// Per-channel sum / sum-of-squares for BN batch stats (block = (c, n))
// ---------------------------------------------------------------------------
__global__ __launch_bounds__(256)
void stats_accum(const unsigned short* __restrict__ t,
                 float* __restrict__ sum, float* __restrict__ sumsq) {
    const int c = blockIdx.x, n = blockIdx.y;
    const unsigned short* p = t + ((long)n * Cc + c) * HWc;
    float s = 0.f, s2 = 0.f;
    for (int i = threadIdx.x; i < HWc; i += 256) {
        float v = bf2f(p[i]); s += v; s2 += v * v;
    }
    __shared__ float rs[256], rq[256];
    rs[threadIdx.x] = s; rq[threadIdx.x] = s2;
    __syncthreads();
    for (int o = 128; o > 0; o >>= 1) {
        if (threadIdx.x < o) {
            rs[threadIdx.x] += rs[threadIdx.x + o];
            rq[threadIdx.x] += rq[threadIdx.x + o];
        }
        __syncthreads();
    }
    if (threadIdx.x == 0) {
        atomicAdd(&sum[c],   rs[0]);
        atomicAdd(&sumsq[c], rq[0]);
    }
}

__global__ void stats_finalize(const float* __restrict__ sum,
                               const float* __restrict__ sumsq,
                               const float* __restrict__ g,
                               const float* __restrict__ be,
                               float* __restrict__ scale,
                               float* __restrict__ shift) {
    int c = blockIdx.x * blockDim.x + threadIdx.x;
    if (c >= Cc) return;
    const float cnt = (float)((long)Nn * HWc);
    float m  = sum[c] / cnt;
    float v  = sumsq[c] / cnt - m * m;
    float sc = g[c] * rsqrtf(v + EPSv);
    scale[c] = sc;
    shift[c] = be[c] - m * sc;
}

// ---------------------------------------------------------------------------
// combine: o1n = shift_up(o2_prev) + bn31(r1raw); o2n = o1n + bn32(r2raw);
// then shuffle: a2 = (c<HALF ? o2n : o1n) -> overwrite r1 buffer;
//               b2 = (c<HALF ? o1n : o2n) -> overwrite r2 buffer.
// o2_prev[c] = c<HALF ? bn11(t1)[c] : bn12(t2)[c]   (read at h+1, zero last row)
// ---------------------------------------------------------------------------
__global__ __launch_bounds__(256)
void combine_kernel(unsigned short* __restrict__ r1, unsigned short* __restrict__ r2,
                    const unsigned short* __restrict__ t1, const unsigned short* __restrict__ t2,
                    const float* __restrict__ sc0, const float* __restrict__ sh0,
                    const float* __restrict__ sc1, const float* __restrict__ sh1,
                    const float* __restrict__ sc2, const float* __restrict__ sh2,
                    const float* __restrict__ sc3, const float* __restrict__ sh3) {
    long i = (long)blockIdx.x * 256 + threadIdx.x;
    if (i >= Etot) return;
    int q = (int)(i % HWc);
    int c = (int)((i / HWc) % Cc);
    int h = q >> 7;
    float rv1 = bf2f(r1[i]) * sc2[c] + sh2[c];
    float rv2 = bf2f(r2[i]) * sc3[c] + sh3[c];
    float f = 0.f;
    if (h < Hh - 1) {
        long j = i + Ww;                                   // (h+1, same n,c)
        f = (c < HALFC) ? bf2f(t1[j]) * sc0[c] + sh0[c]
                        : bf2f(t2[j]) * sc1[c] + sh1[c];
    }
    float o1n = f + rv1;
    float o2n = o1n + rv2;
    r1[i] = f2bf(c < HALFC ? o2n : o1n);   // a2 (in place, same-index RW)
    r2[i] = f2bf(c < HALFC ? o1n : o2n);   // b2
}

// ---------------------------------------------------------------------------
// final: out = bn(u_raw) + x   (f32 output)
// ---------------------------------------------------------------------------
__global__ __launch_bounds__(256)
void final_kernel(const unsigned short* __restrict__ u,
                  const float* __restrict__ sc, const float* __restrict__ sh,
                  const float* __restrict__ x, float* __restrict__ out) {
    long i = (long)blockIdx.x * 256 + threadIdx.x;
    if (i >= Etot) return;
    int c = (int)((i / HWc) % Cc);
    out[i] = bf2f(u[i]) * sc[c] + sh[c] + x[i];
}

// ---------------------------------------------------------------------------
// Host orchestration. Workspace layout (needs ~309 MB):
//   [Ab | Bb | Cb | Db] 4 bf16 tensors of Etot + stats (24*C f32) + bf16 weights
// ---------------------------------------------------------------------------
extern "C" void kernel_launch(void* const* d_in, const int* in_sizes, int n_in,
                              void* d_out, int out_size, void* d_ws, size_t ws_size,
                              hipStream_t stream) {
    (void)in_sizes; (void)n_in; (void)out_size; (void)ws_size;

    const float* x1  = (const float*)d_in[0];
    const float* x2  = (const float*)d_in[1];
    const float* w11 = (const float*)d_in[2];
    const float* g11 = (const float*)d_in[4];
    const float* be11= (const float*)d_in[5];
    const float* w12 = (const float*)d_in[6];
    const float* g12 = (const float*)d_in[8];
    const float* be12= (const float*)d_in[9];
    const float* w31 = (const float*)d_in[10];
    const float* g31 = (const float*)d_in[12];
    const float* be31= (const float*)d_in[13];
    const float* w32 = (const float*)d_in[14];
    const float* g32 = (const float*)d_in[16];
    const float* be32= (const float*)d_in[17];
    const float* w21 = (const float*)d_in[18];
    const float* g21 = (const float*)d_in[20];
    const float* be21= (const float*)d_in[21];
    const float* w22 = (const float*)d_in[22];
    const float* g22 = (const float*)d_in[24];
    const float* be22= (const float*)d_in[25];
    float* out = (float*)d_out;

    unsigned short* Ab = (unsigned short*)d_ws;
    unsigned short* Bb = Ab + Etot;
    unsigned short* Cb = Bb + Etot;
    unsigned short* Db = Cb + Etot;
    float* stats = (float*)(Db + Etot);          // 6 convs * {sum,sumsq,scale,shift}[C]
    auto SUM = [&](int j){ return stats + (j*4+0)*Cc; };
    auto SQ  = [&](int j){ return stats + (j*4+1)*Cc; };
    auto SCL = [&](int j){ return stats + (j*4+2)*Cc; };
    auto SHF = [&](int j){ return stats + (j*4+3)*Cc; };
    unsigned short* wb11 = (unsigned short*)(stats + 24*Cc);
    unsigned short* wb12 = wb11 + Cc*Cc;
    unsigned short* wb21 = wb12 + Cc*Cc;
    unsigned short* wb22 = wb21 + Cc*Cc;
    unsigned short* wb31 = wb22 + Cc*Cc;
    unsigned short* wb32 = wb31 + (long)Cc*Cc*9;

    hipMemsetAsync(stats, 0, 24*Cc*sizeof(float), stream);

    // bf16 conversions (inputs + weights; biases cancel under batch-norm)
    cvt_f32_bf16<<<1024,256,0,stream>>>(x1, Ab, Etot);
    cvt_f32_bf16<<<1024,256,0,stream>>>(x2, Bb, Etot);
    cvt_f32_bf16<<<256,256,0,stream>>>(w11, wb11, (long)Cc*Cc);
    cvt_f32_bf16<<<256,256,0,stream>>>(w12, wb12, (long)Cc*Cc);
    cvt_f32_bf16<<<256,256,0,stream>>>(w21, wb21, (long)Cc*Cc);
    cvt_f32_bf16<<<256,256,0,stream>>>(w22, wb22, (long)Cc*Cc);
    cvt_f32_bf16<<<512,256,0,stream>>>(w31, wb31, (long)Cc*Cc*9);
    cvt_f32_bf16<<<512,256,0,stream>>>(w32, wb32, (long)Cc*Cc*9);

    const dim3 gg(Nn*HWc/BNT, Cc/BM);     // (768, 3)
    const dim3 gs(Cc, Nn);                // stats grid
    const int  ge = (int)(Etot/256);      // elementwise grid

    // stage 1: 1x1 convs (raw)
    gemm_conv<1,false><<<gg,256,0,stream>>>(Ab,Ab,nullptr,nullptr,nullptr,nullptr, wb11, Cb);
    gemm_conv<1,false><<<gg,256,0,stream>>>(Bb,Bb,nullptr,nullptr,nullptr,nullptr, wb12, Db);
    stats_accum<<<gs,256,0,stream>>>(Cb, SUM(0), SQ(0));
    stats_accum<<<gs,256,0,stream>>>(Db, SUM(1), SQ(1));
    stats_finalize<<<3,128,0,stream>>>(SUM(0),SQ(0), g11,be11, SCL(0),SHF(0));
    stats_finalize<<<3,128,0,stream>>>(SUM(1),SQ(1), g12,be12, SCL(1),SHF(1));

    // stage 2: 3x3 convs; BN + channel shuffle folded into activation fetch
    gemm_conv<3,true><<<gg,256,0,stream>>>(Db,Cb, SCL(1),SHF(1), SCL(0),SHF(0), wb31, Ab);
    gemm_conv<3,true><<<gg,256,0,stream>>>(Cb,Db, SCL(0),SHF(0), SCL(1),SHF(1), wb32, Bb);
    stats_accum<<<gs,256,0,stream>>>(Ab, SUM(2), SQ(2));
    stats_accum<<<gs,256,0,stream>>>(Bb, SUM(3), SQ(3));
    stats_finalize<<<3,128,0,stream>>>(SUM(2),SQ(2), g31,be31, SCL(2),SHF(2));
    stats_finalize<<<3,128,0,stream>>>(SUM(3),SQ(3), g32,be32, SCL(3),SHF(3));

    // shift + add + second shuffle (in place over Ab/Bb)
    combine_kernel<<<ge,256,0,stream>>>(Ab,Bb, Cb,Db,
        SCL(0),SHF(0), SCL(1),SHF(1), SCL(2),SHF(2), SCL(3),SHF(3));

    // stage 3: final 1x1 convs (raw)
    gemm_conv<1,false><<<gg,256,0,stream>>>(Ab,Ab,nullptr,nullptr,nullptr,nullptr, wb21, Cb);
    gemm_conv<1,false><<<gg,256,0,stream>>>(Bb,Bb,nullptr,nullptr,nullptr,nullptr, wb22, Db);
    stats_accum<<<gs,256,0,stream>>>(Cb, SUM(4), SQ(4));
    stats_accum<<<gs,256,0,stream>>>(Db, SUM(5), SQ(5));
    stats_finalize<<<3,128,0,stream>>>(SUM(4),SQ(4), g21,be21, SCL(4),SHF(4));
    stats_finalize<<<3,128,0,stream>>>(SUM(5),SQ(5), g22,be22, SCL(5),SHF(5));

    // BN + residual -> f32 outputs
    final_kernel<<<ge,256,0,stream>>>(Cb, SCL(4),SHF(4), x1, out);
    final_kernel<<<ge,256,0,stream>>>(Db, SCL(5),SHF(5), x2, out + Etot);
}
